// GNNModel_80513456930926
// MI455X (gfx1250) — matrix operand order repaired and checked by reading, not compiled, not run on hardware
//
#include <hip/hip_runtime.h>
#include <hip/hip_bf16.h>

// ---------------------------------------------------------------------------
// SR-GNN forward for MI455X (gfx1250). Heavy GEMMs run through
// v_wmma_f32_16x16x32_bf16 with global->LDS staging done by the CDNA5 async
// engine (global_load_async_to_lds_b128 + s_wait_asynccnt), double-buffered
// so each K-step overlaps the next tile's DMA with the current tile's WMMAs.
// Workload is HBM-bandwidth bound (~2.5 GB traffic, 819 MB of it the final
// score matrix), so bf16 operand compression + fused bias epilogues are the
// main wins; the WMMA GEMM uses 32x32 per-wave register blocking (4 wmma per
// 8 LDS fragment loads).
// ---------------------------------------------------------------------------

typedef __attribute__((ext_vector_type(16))) __bf16 v16bf;
typedef __attribute__((ext_vector_type(8)))  float  v8f;

#define Hh 256
#define NUM_GRAPHS 4096
#define NODES_PER_G 20
#define EDGES_PER_G 40
#define NTOT (NUM_GRAPHS * NODES_PER_G)   // 81920
#define ETOT (NUM_GRAPHS * EDGES_PER_G)   // 163840
#define N_NODE 50000

__device__ __forceinline__ unsigned short f2bf(float f) {
    unsigned int u = __float_as_uint(f);
    unsigned int r = (u + 0x7FFFu + ((u >> 16) & 1u)) >> 16;   // RNE
    return (unsigned short)r;
}
__device__ __forceinline__ float sigm(float x) { return 1.0f / (1.0f + __expf(-x)); }

// ------------------------------ elementwise --------------------------------

__global__ void k_convert_bf16(const float* __restrict__ in,
                               unsigned short* __restrict__ out, long long n) {
    long long i = (long long)blockIdx.x * blockDim.x + threadIdx.x;
    if (i < n) out[i] = f2bf(in[i]);
}

// gated_w[l] is used as h @ gw[l]; store B^T form: out[l][n][k] = in[l][k][n]
__global__ void k_transpose_convert(const float* __restrict__ in,
                                    unsigned short* __restrict__ out, int L) {
    long long i = (long long)blockIdx.x * blockDim.x + threadIdx.x;
    long long tot = (long long)L * Hh * Hh;
    if (i >= tot) return;
    int l = (int)(i / (Hh * Hh));
    int rem = (int)(i % (Hh * Hh));
    int n = rem / Hh, k = rem % Hh;
    out[i] = f2bf(in[(long long)l * Hh * Hh + (long long)k * Hh + n]);
}

__global__ void k_gather(const int* __restrict__ x, const float* __restrict__ emb,
                         float* __restrict__ h, unsigned short* __restrict__ hbf) {
    long long i = (long long)blockIdx.x * blockDim.x + threadIdx.x;
    if (i >= (long long)NTOT * Hh) return;
    int n = (int)(i >> 8), c = (int)(i & 255);
    float v = emb[(long long)(x[n] - 1) * Hh + c];
    h[i] = v; hbf[i] = f2bf(v);
}

__global__ void k_scatter_add(const float* __restrict__ m, const int* __restrict__ ei,
                              float* __restrict__ agg) {
    long long i = (long long)blockIdx.x * blockDim.x + threadIdx.x;
    if (i >= (long long)ETOT * Hh) return;
    int e = (int)(i >> 8), c = (int)(i & 255);
    int s = ei[e], d = ei[ETOT + e];
    atomicAdd(&agg[(long long)d * Hh + c], m[(long long)s * Hh + c]);
}

__global__ void k_gru(const float* __restrict__ gi, const float* __restrict__ gh,
                      float* __restrict__ h, unsigned short* __restrict__ hbf) {
    long long i = (long long)blockIdx.x * blockDim.x + threadIdx.x;
    if (i >= (long long)NTOT * Hh) return;
    int n = (int)(i >> 8), c = (int)(i & 255);
    long long b = (long long)n * (3 * Hh) + c;
    float ir = gi[b], iz = gi[b + Hh], inn = gi[b + 2 * Hh];
    float hr = gh[b], hz = gh[b + Hh], hnn = gh[b + 2 * Hh];
    float r = sigm(ir + hr);
    float z = sigm(iz + hz);
    float nn = tanhf(inn + r * hnn);
    float hv = (1.0f - z) * nn + z * h[i];
    h[i] = hv; hbf[i] = f2bf(hv);
}

// v_n = h[last node of each graph]; also drops v_n into s_cat[:, :H]
__global__ void k_vn(const float* __restrict__ h, float* __restrict__ vn,
                     unsigned short* __restrict__ vnbf, float* __restrict__ scat) {
    long long i = (long long)blockIdx.x * blockDim.x + threadIdx.x;
    if (i >= (long long)NUM_GRAPHS * Hh) return;
    int b = (int)(i >> 8), c = (int)(i & 255);
    float v = h[((long long)b * NODES_PER_G + (NODES_PER_G - 1)) * Hh + c];
    vn[i] = v; vnbf[i] = f2bf(v);
    scat[(long long)b * (2 * Hh) + c] = v;
}

// alpha[n] = sigmoid(t2[n]+t1[n/20]+b1+b2) . qw + qb    (one wave per node)
__global__ void k_alpha(const float* __restrict__ t2, const float* __restrict__ t1,
                        const float* __restrict__ b1, const float* __restrict__ b2,
                        const float* __restrict__ qw, const float* __restrict__ qb,
                        float* __restrict__ alpha) {
    int wave = (int)(((long long)blockIdx.x * blockDim.x + threadIdx.x) >> 5);
    int lane = threadIdx.x & 31;
    if (wave >= NTOT) return;
    int g = wave / NODES_PER_G;
    float a = 0.0f;
#pragma unroll
    for (int i = 0; i < Hh / 32; ++i) {
        int c = lane + 32 * i;
        float v = t2[(long long)wave * Hh + c] + t1[(long long)g * Hh + c] + b1[c] + b2[c];
        a += sigm(v) * qw[c];
    }
#pragma unroll
    for (int off = 16; off > 0; off >>= 1) a += __shfl_down(a, off, 32);
    if (lane == 0) alpha[wave] = a + qb[0];
}

__global__ void k_sg(const float* __restrict__ alpha, const float* __restrict__ h,
                     float* __restrict__ sg) {
    long long i = (long long)blockIdx.x * blockDim.x + threadIdx.x;
    if (i >= (long long)NTOT * Hh) return;
    int n = (int)(i >> 8), c = (int)(i & 255);
    atomicAdd(&sg[(long long)(n / NODES_PER_G) * Hh + c], alpha[n] * h[i]);
}

__global__ void k_cat2(const float* __restrict__ sg, float* __restrict__ scat) {
    long long i = (long long)blockIdx.x * blockDim.x + threadIdx.x;
    if (i >= (long long)NUM_GRAPHS * Hh) return;
    int b = (int)(i >> 8), c = (int)(i & 255);
    scat[(long long)b * (2 * Hh) + Hh + c] = sg[i];
}

// --------------------------- WMMA bf16 GEMM --------------------------------
// C[M,N] (f32) = A[M,K] (bf16, row-major) * B[N,K]^T (bf16, row-major) + bias
// Block: 256 threads = 8 waves (2x4 wave grid); each wave computes a 32x32 C
// tile (2x2 of 16x16 wmma tiles) -> block tile 64(M) x 128(N), K stepped 32.
// Global->LDS staging uses the gfx1250 async engine with double buffering:
// one s_wait_asynccnt + barrier per K-step; next tile's DMA overlaps WMMAs.
// Fragment packing per CDNA5 ISA 7.12.2 (halves are LDS-contiguous -> the
// 8 dword reads fold into 2x ds_load_b128 per fragment).

#define TM 64
#define TN 128
#define TK 32
#define LDSS 40   // padded bf16 row stride (80 B, multiple of 16 for b128)

__device__ __forceinline__ void async_tile(const unsigned short* __restrict__ A,
                                           const unsigned short* __restrict__ B,
                                           unsigned short* As, unsigned short* Bs,
                                           int t, int m0, int n0,
                                           int M, int N, int K, int k0) {
    int row = t >> 2;            // 0..63
    int col = (t & 3) << 3;      // 0,8,16,24  (8 bf16 = 16 B per async op)
    {   // A tile: 64 x 32
        int r = m0 + row; if (r >= M) r = M - 1;       // clamp: junk rows are
        unsigned lds = (unsigned)(unsigned long long)&As[row * LDSS + col];
        unsigned long long g = (unsigned long long)(A + (size_t)r * K + k0 + col);
        asm volatile("global_load_async_to_lds_b128 %0, %1, off"
                     :: "v"(lds), "v"(g) : "memory");
    }
#pragma unroll
    for (int j = 0; j < 2; ++j) {   // B tile: 128 x 32
        int br = row + 64 * j;
        int r = n0 + br; if (r >= N) r = N - 1;        // never stored anyway
        unsigned lds = (unsigned)(unsigned long long)&Bs[br * LDSS + col];
        unsigned long long g = (unsigned long long)(B + (size_t)r * K + k0 + col);
        asm volatile("global_load_async_to_lds_b128 %0, %1, off"
                     :: "v"(lds), "v"(g) : "memory");
    }
}

__global__ __launch_bounds__(256)
void k_gemm_bf16(const unsigned short* __restrict__ A,
                 const unsigned short* __restrict__ B,
                 const float* __restrict__ bias,
                 float* __restrict__ C, int M, int N, int K) {
    __shared__ unsigned short As[2][TM * LDSS];
    __shared__ unsigned short Bs[2][TN * LDSS];

    int m0 = blockIdx.y * TM;
    int n0 = blockIdx.x * TN;
    int t = threadIdx.x;
    int lane = t & 31, w = t >> 5;
    int wm = w >> 2, wn = w & 3;

    v8f acc[2][2] = {};
    int nk = K / TK;

    async_tile(A, B, As[0], Bs[0], t, m0, n0, M, N, K, 0);

    for (int kt = 0; kt < nk; ++kt) {
        asm volatile("s_wait_asynccnt 0x0" ::: "memory");
        __syncthreads();
        if (kt + 1 < nk)
            async_tile(A, B, As[(kt + 1) & 1], Bs[(kt + 1) & 1],
                       t, m0, n0, M, N, K, (kt + 1) * TK);

        const unsigned short* as = As[kt & 1];
        const unsigned short* bs = Bs[kt & 1];
        union { v16bf v; unsigned int u[8]; } af[2], bfr[2];
        int hiA = (lane < 16) ? 0 : 8;
        int hiB = (lane < 16) ? 0 : 8;
#pragma unroll
        for (int mi = 0; mi < 2; ++mi) {
            int arow = wm * 32 + mi * 16 + (lane & 15);
#pragma unroll
            for (int g = 0; g < 8; ++g) {
                int ka = ((g < 4) ? 2 * g : 2 * g + 8) + hiA;
                af[mi].u[g] = *(const unsigned int*)&as[arow * LDSS + ka];
            }
        }
#pragma unroll
        for (int ni = 0; ni < 2; ++ni) {
            int brow = wn * 32 + ni * 16 + (lane & 15);
#pragma unroll
            for (int g = 0; g < 8; ++g) {
                bfr[ni].u[g] = *(const unsigned int*)&bs[brow * LDSS + ((g < 4) ? 2 * g : 2 * g + 8) + hiB];
            }
        }
#pragma unroll
        for (int mi = 0; mi < 2; ++mi)
#pragma unroll
            for (int ni = 0; ni < 2; ++ni)
                acc[mi][ni] = __builtin_amdgcn_wmma_f32_16x16x32_bf16(
                    false, af[mi].v, false, bfr[ni].v, (short)0,
                    acc[mi][ni], false, false);
    }

#pragma unroll
    for (int mi = 0; mi < 2; ++mi) {
        int rbase = m0 + wm * 32 + mi * 16 + ((lane < 16) ? 0 : 8);
#pragma unroll
        for (int ni = 0; ni < 2; ++ni) {
            int ccol = n0 + wn * 32 + ni * 16 + (lane & 15);
            if (ccol < N) {
#pragma unroll
                for (int r = 0; r < 8; ++r) {
                    int rr = rbase + r;
                    if (rr < M) {
                        float v = acc[mi][ni][r];
                        if (bias) v += bias[ccol];
                        C[(size_t)rr * N + ccol] = v;
                    }
                }
            }
        }
    }
}

// ------------------------------- driver ------------------------------------

static inline void* carve(char*& p, size_t bytes) {
    void* r = (void*)p;
    p += (bytes + 255) & ~size_t(255);
    return r;
}

static inline void gemm(const unsigned short* A, const unsigned short* B,
                        const float* bias, float* C, int M, int N, int K,
                        hipStream_t s) {
    dim3 grid((N + TN - 1) / TN, (M + TM - 1) / TM);
    k_gemm_bf16<<<grid, 256, 0, s>>>(A, B, bias, C, M, N, K);
}

extern "C" void kernel_launch(void* const* d_in, const int* in_sizes, int n_in,
                              void* d_out, int out_size, void* d_ws, size_t ws_size,
                              hipStream_t stream) {
    const int*   x    = (const int*)d_in[0];
    const int*   ei   = (const int*)d_in[1];
    // d_in[2] = batch (uniform: n/20, unused)
    const float* emb  = (const float*)d_in[3];
    const float* gw   = (const float*)d_in[4];
    const float* wih  = (const float*)d_in[5];
    const float* whh  = (const float*)d_in[6];
    const float* bih  = (const float*)d_in[7];
    const float* bhh  = (const float*)d_in[8];
    const float* W1   = (const float*)d_in[9];
    const float* b1   = (const float*)d_in[10];
    const float* W2   = (const float*)d_in[11];
    const float* b2   = (const float*)d_in[12];
    const float* qw   = (const float*)d_in[13];
    const float* qb   = (const float*)d_in[14];
    const float* W3   = (const float*)d_in[15];
    const float* b3   = (const float*)d_in[16];

    char* p = (char*)d_ws;
    float* h      = (float*)carve(p, (size_t)NTOT * Hh * 4);
    unsigned short* hbf   = (unsigned short*)carve(p, (size_t)NTOT * Hh * 2);
    float* m      = (float*)carve(p, (size_t)NTOT * Hh * 4);          // also t2
    float* agg    = (float*)carve(p, (size_t)NTOT * Hh * 4);
    unsigned short* aggbf = (unsigned short*)carve(p, (size_t)NTOT * Hh * 2);
    float* gi     = (float*)carve(p, (size_t)NTOT * 3 * Hh * 4);
    float* gh     = (float*)carve(p, (size_t)NTOT * 3 * Hh * 4);
    unsigned short* gwt   = (unsigned short*)carve(p, (size_t)2 * Hh * Hh * 2);
    unsigned short* wihbf = (unsigned short*)carve(p, (size_t)3 * Hh * Hh * 2);
    unsigned short* whhbf = (unsigned short*)carve(p, (size_t)3 * Hh * Hh * 2);
    unsigned short* W1bf  = (unsigned short*)carve(p, (size_t)Hh * Hh * 2);
    unsigned short* W2bf  = (unsigned short*)carve(p, (size_t)Hh * Hh * 2);
    unsigned short* W3bf  = (unsigned short*)carve(p, (size_t)Hh * 2 * Hh * 2);
    unsigned short* embbf = (unsigned short*)carve(p, (size_t)N_NODE * Hh * 2);
    float* vn     = (float*)carve(p, (size_t)NUM_GRAPHS * Hh * 4);
    unsigned short* vnbf  = (unsigned short*)carve(p, (size_t)NUM_GRAPHS * Hh * 2);
    float* t1     = (float*)carve(p, (size_t)NUM_GRAPHS * Hh * 4);
    float* alpha  = (float*)carve(p, (size_t)NTOT * 4);
    float* sg     = (float*)carve(p, (size_t)NUM_GRAPHS * Hh * 4);
    float* scat   = (float*)carve(p, (size_t)NUM_GRAPHS * 2 * Hh * 4);
    unsigned short* scatbf = (unsigned short*)carve(p, (size_t)NUM_GRAPHS * 2 * Hh * 2);
    float* sh     = (float*)carve(p, (size_t)NUM_GRAPHS * Hh * 4);
    unsigned short* shbf  = (unsigned short*)carve(p, (size_t)NUM_GRAPHS * Hh * 2);

    auto blocks = [](long long n) { return (unsigned)((n + 255) / 256); };

    // ---- weight prep (bf16, B^T form where needed) ----
    k_transpose_convert<<<blocks(2LL * Hh * Hh), 256, 0, stream>>>(gw, gwt, 2);
    k_convert_bf16<<<blocks(3LL * Hh * Hh), 256, 0, stream>>>(wih, wihbf, 3LL * Hh * Hh);
    k_convert_bf16<<<blocks(3LL * Hh * Hh), 256, 0, stream>>>(whh, whhbf, 3LL * Hh * Hh);
    k_convert_bf16<<<blocks((long long)Hh * Hh), 256, 0, stream>>>(W1, W1bf, (long long)Hh * Hh);
    k_convert_bf16<<<blocks((long long)Hh * Hh), 256, 0, stream>>>(W2, W2bf, (long long)Hh * Hh);
    k_convert_bf16<<<blocks(2LL * Hh * Hh), 256, 0, stream>>>(W3, W3bf, 2LL * Hh * Hh);
    k_convert_bf16<<<blocks((long long)N_NODE * Hh), 256, 0, stream>>>(emb, embbf, (long long)N_NODE * Hh);

    // ---- embedding gather ----
    k_gather<<<blocks((long long)NTOT * Hh), 256, 0, stream>>>(x, emb, h, hbf);

    // ---- 2x GatedGraphConv ----
    for (int l = 0; l < 2; ++l) {
        gemm(hbf, gwt + (size_t)l * Hh * Hh, nullptr, m, NTOT, Hh, Hh, stream);
        hipMemsetAsync(agg, 0, (size_t)NTOT * Hh * 4, stream);
        k_scatter_add<<<blocks((long long)ETOT * Hh), 256, 0, stream>>>(m, ei, agg);
        k_convert_bf16<<<blocks((long long)NTOT * Hh), 256, 0, stream>>>(agg, aggbf, (long long)NTOT * Hh);
        gemm(aggbf, wihbf, bih, gi, NTOT, 3 * Hh, Hh, stream);
        gemm(hbf,   whhbf, bhh, gh, NTOT, 3 * Hh, Hh, stream);
        k_gru<<<blocks((long long)NTOT * Hh), 256, 0, stream>>>(gi, gh, h, hbf);
    }

    // ---- attention readout ----
    gemm(hbf, W2bf, nullptr, m, NTOT, Hh, Hh, stream);                 // t2 = h @ W2^T
    k_vn<<<blocks((long long)NUM_GRAPHS * Hh), 256, 0, stream>>>(h, vn, vnbf, scat);
    gemm(vnbf, W1bf, nullptr, t1, NUM_GRAPHS, Hh, Hh, stream);         // t1 = v_n @ W1^T
    k_alpha<<<(NTOT * 32) / 256, 256, 0, stream>>>(m, t1, b1, b2, qw, qb, alpha);
    hipMemsetAsync(sg, 0, (size_t)NUM_GRAPHS * Hh * 4, stream);
    k_sg<<<blocks((long long)NTOT * Hh), 256, 0, stream>>>(alpha, h, sg);
    k_cat2<<<blocks((long long)NUM_GRAPHS * Hh), 256, 0, stream>>>(sg, scat);
    k_convert_bf16<<<blocks((long long)NUM_GRAPHS * 2 * Hh), 256, 0, stream>>>(
        scat, scatbf, (long long)NUM_GRAPHS * 2 * Hh);
    gemm(scatbf, W3bf, b3, sh, NUM_GRAPHS, Hh, 2 * Hh, stream);        // s_h
    k_convert_bf16<<<blocks((long long)NUM_GRAPHS * Hh), 256, 0, stream>>>(
        sh, shbf, (long long)NUM_GRAPHS * Hh);

    // ---- final scores: [4096,256] x [50000,256]^T -> 819 MB of f32 ----
    gemm(shbf, embbf, nullptr, (float*)d_out, NUM_GRAPHS, N_NODE, Hh, stream);
}
